// CausalSelfAttention_91147795956143
// MI455X (gfx1250) — matrix operand order repaired and checked by reading, not compiled, not run on hardware
//
#include <hip/hip_runtime.h>

typedef __bf16 bf16;
typedef __attribute__((ext_vector_type(16))) __bf16 v16bf;
typedef __attribute__((ext_vector_type(8)))  __bf16 v8bf;
typedef __attribute__((ext_vector_type(8)))  float  v8f;

#define BATCH 4
#define SEQ   2048
#define DIM   1024
#define HEADS 16
#define HD    64
#define NQKV  (3*DIM)
#define SM_SCALE 0.125f   // HD^-0.5

// ---------------------------------------------------------------------------
// WMMA helper: D = A(16x32 bf16) * B(32x16 bf16) + C(16x16 f32)
// ---------------------------------------------------------------------------
__device__ __forceinline__ v8f wmma_bf16(v16bf a, v16bf b, v8f c) {
  return __builtin_amdgcn_wmma_f32_16x16x32_bf16(
      /*neg_a=*/false, a, /*neg_b=*/false, b,
      /*c_mod=*/(short)0, c, /*reuse_a=*/false, /*reuse_b=*/false);
}

// A fragment (16x32, M x K) from row-major storage, leading dim ld (elements).
// ISA layout: lane = m + 16*khalf; lane holds K = {8h..8h+7} U {16+8h..16+8h+7}.
__device__ __forceinline__ v16bf load_frag_a(const bf16* base, int ld) {
  const int lane = threadIdx.x & 31;
  const int m = lane & 15, h = lane >> 4;
  const bf16* p = base + (size_t)m * ld + 8 * h;
  v8bf lo = *(const v8bf*)(p);        // k = 8h .. 8h+7
  v8bf hi = *(const v8bf*)(p + 16);   // k = 16+8h .. 16+8h+7
  v16bf r;
#pragma unroll
  for (int i = 0; i < 8; ++i) { r[i] = lo[i]; r[i + 8] = hi[i]; }
  return r;
}

// B fragment (32x16, K x N) loaded from K-major storage: baseT[n*ld + k].
// ISA layout: lane = n + 16*khalf; lane holds K = {8h..8h+7} U {16+8h..16+8h+7}.
__device__ __forceinline__ v16bf load_frag_b(const bf16* baseT, int ld) {
  const int lane = threadIdx.x & 31;
  const int n = lane & 15, g = lane >> 4;
  const bf16* p = baseT + (size_t)n * ld + 8 * g;
  v8bf lo = *(const v8bf*)(p);
  v8bf hi = *(const v8bf*)(p + 16);
  v16bf r;
#pragma unroll
  for (int i = 0; i < 8; ++i) { r[i] = lo[i]; r[i + 8] = hi[i]; }
  return r;
}

// ---------------------------------------------------------------------------
// Prep: f32 -> bf16 convert (row-major copy)
// ---------------------------------------------------------------------------
__global__ void k_cvt(const float* __restrict__ in, bf16* __restrict__ out, int n) {
  int i = blockIdx.x * blockDim.x + threadIdx.x;
  if (i < n) out[i] = (bf16)in[i];
}

// f32 (rows x cols, row-major) -> bf16 transposed (cols x rows)
__global__ void k_cvt_t(const float* __restrict__ in, bf16* __restrict__ out,
                        int rows, int cols) {
  int i = blockIdx.x * blockDim.x + threadIdx.x;
  if (i < rows * cols) {
    int k = i / cols, n = i - k * cols;
    out[(size_t)n * rows + k] = (bf16)in[i];
  }
}

// ---------------------------------------------------------------------------
// Kernel 1: qkv = x @ w_qkv  (bf16 WMMA, f32 acc) + fused RoPE + scatter.
// Each wave: 16 rows x 64 cols (= one head slice of q/k/v).
// grid = (M/64, NQKV/64), block = 128 (4 waves along M).
// ---------------------------------------------------------------------------
__global__ __launch_bounds__(128) void
k_qkv(const bf16* __restrict__ xb, const bf16* __restrict__ wT,
      float* __restrict__ out_k, float* __restrict__ out_v,
      bf16* __restrict__ q_ws, bf16* __restrict__ k_ws, bf16* __restrict__ v_wsT) {
  const int lane = threadIdx.x & 31;
  const int wave = threadIdx.x >> 5;
  const int m0 = blockIdx.x * 64 + wave * 16;
  const int n0 = blockIdx.y * 64;

  v8f acc[4];
#pragma unroll
  for (int t = 0; t < 4; ++t) acc[t] = (v8f)(0.0f);

  for (int k = 0; k < DIM; k += 32) {
    v16bf a = load_frag_a(xb + (size_t)m0 * DIM + k, DIM);
#pragma unroll
    for (int t = 0; t < 4; ++t) {
      v16bf b = load_frag_b(wT + (size_t)(n0 + 16 * t) * DIM + k, DIM);
      acc[t] = wmma_bf16(a, b, acc[t]);
    }
  }

  const int half = lane >> 4, nl = lane & 15;
  const int sec = n0 / DIM;           // 0=q 1=k 2=v
  const int hh  = (n0 % DIM) / HD;    // head index

  if (sec == 2) {                     // ---- V: no RoPE
#pragma unroll
    for (int t = 0; t < 4; ++t) {
      int d = 16 * t + nl;
#pragma unroll
      for (int vv = 0; vv < 8; ++vv) {
        int m = m0 + 8 * half + vv;
        int b = m / SEQ, s = m - b * SEQ;
        float val = acc[t][vv];
        size_t bh = (size_t)(b * HEADS + hh);
        out_v[(bh * SEQ + s) * HD + d] = val;
        v_wsT[(bh * HD + d) * SEQ + s] = (bf16)val;   // d-major for PV B-operand
      }
    }
  } else {                            // ---- Q/K: RoPE on pairs (d, d+32)
#pragma unroll
    for (int t = 0; t < 2; ++t) {
      int d = 16 * t + nl;            // d in [0,32)
      float inv_freq = __powf(10000.0f, -(float)d * (1.0f / 32.0f));
#pragma unroll
      for (int vv = 0; vv < 8; ++vv) {
        int m = m0 + 8 * half + vv;
        int b = m / SEQ, s = m - b * SEQ;
        float ang = (float)s * inv_freq, sn, cs;
        __sincosf(ang, &sn, &cs);
        float x0 = acc[t][vv], x2 = acc[t + 2][vv];
        float r0 = x0 * cs - x2 * sn;
        float r2 = x2 * cs + x0 * sn;
        size_t bh = (size_t)(b * HEADS + hh);
        size_t i0 = (bh * SEQ + s) * HD + d;
        if (sec == 0) {
          q_ws[i0] = (bf16)r0;  q_ws[i0 + 32] = (bf16)r2;
        } else {
          out_k[i0] = r0;       out_k[i0 + 32] = r2;
          k_ws[i0] = (bf16)r0;  k_ws[i0 + 32] = (bf16)r2;
        }
      }
    }
  }
}

// ---------------------------------------------------------------------------
// Kernel 2: causal flash attention. One wave per 16-query tile per (b,h).
// grid = (SEQ/64, BATCH*HEADS), block = 128 (4 waves = 4 q-tiles).
// ---------------------------------------------------------------------------
__global__ __launch_bounds__(128) void
k_attn(const bf16* __restrict__ q_ws, const bf16* __restrict__ k_ws,
       const bf16* __restrict__ v_wsT, bf16* __restrict__ attn_ws) {
  __shared__ __align__(16) bf16 pbuf[4][16 * 32];   // per-wave P staging

  const int lane = threadIdx.x & 31;
  const int wave = threadIdx.x >> 5;
  const int half = lane >> 4, nl = lane & 15;
  const int qt = blockIdx.x * 4 + wave;             // 0..127
  const int bh = blockIdx.y;
  const int b = bh >> 4, hh = bh & 15;
  const int q0 = qt * 16;

  const bf16* qbase = q_ws  + (size_t)bh * SEQ * HD;
  const bf16* kbase = k_ws  + (size_t)bh * SEQ * HD;
  const bf16* vbase = v_wsT + (size_t)bh * HD * SEQ;

  // Q A-fragments for d = 0..31 and 32..63 (loaded once)
  v16bf qa0 = load_frag_a(qbase + (size_t)q0 * HD, HD);
  v16bf qa1 = load_frag_a(qbase + (size_t)q0 * HD + 32, HD);

  v8f o[4];
#pragma unroll
  for (int t = 0; t < 4; ++t) o[t] = (v8f)(0.0f);
  float mrow[8], lrow[8];
#pragma unroll
  for (int vv = 0; vv < 8; ++vv) { mrow[vv] = -1e30f; lrow[vv] = 0.0f; }

  const int kb_end = (q0 + 15) >> 5;
  for (int kb = 0; kb <= kb_end; ++kb) {
    const int key0 = kb * 32;

    // scores: S = Q(16x64) @ K^T(64x32), two 16-key sub-tiles
    v8f s[2];
#pragma unroll
    for (int t = 0; t < 2; ++t) {
      v8f c = (v8f)(0.0f);
      c = wmma_bf16(qa0, load_frag_b(kbase + (size_t)(key0 + 16 * t) * HD,      HD), c);
      c = wmma_bf16(qa1, load_frag_b(kbase + (size_t)(key0 + 16 * t) * HD + 32, HD), c);
      s[t] = c;
    }

    // online softmax (rows live on 16-lane groups, vgpr = m%8)
#pragma unroll
    for (int vv = 0; vv < 8; ++vv) {
      int qrow = q0 + 8 * half + vv;
      float x0 = s[0][vv] * SM_SCALE;
      float x1 = s[1][vv] * SM_SCALE;
      if (key0 + nl      > qrow) x0 = -1e30f;   // causal mask
      if (key0 + 16 + nl > qrow) x1 = -1e30f;
      float mx = fmaxf(x0, x1);
#pragma unroll
      for (int off = 1; off < 16; off <<= 1) mx = fmaxf(mx, __shfl_xor(mx, off, 32));
      float mnew = fmaxf(mrow[vv], mx);
      float r  = __expf(mrow[vv] - mnew);
      float p0 = __expf(x0 - mnew);
      float p1 = __expf(x1 - mnew);
      float ps = p0 + p1;
#pragma unroll
      for (int off = 1; off < 16; off <<= 1) ps += __shfl_xor(ps, off, 32);
      lrow[vv] = lrow[vv] * r + ps;
      mrow[vv] = mnew;
#pragma unroll
      for (int t2 = 0; t2 < 4; ++t2) o[t2][vv] *= r;
      // C-layout -> LDS (row-major 16x32) for A-layout reload
      pbuf[wave][(8 * half + vv) * 32 + nl]      = (bf16)p0;
      pbuf[wave][(8 * half + vv) * 32 + 16 + nl] = (bf16)p1;
    }

    // O += P(16x32) @ V(32x64)
    v16bf pa = load_frag_a(&pbuf[wave][0], 32);
#pragma unroll
    for (int tn = 0; tn < 4; ++tn) {
      v16bf vb = load_frag_b(vbase + (size_t)(16 * tn) * SEQ + key0, SEQ);
      o[tn] = wmma_bf16(pa, vb, o[tn]);
    }
  }

  // epilogue: normalize and write (B,S,D) bf16 for output projection
#pragma unroll
  for (int t = 0; t < 4; ++t) {
    int d = 16 * t + nl;
#pragma unroll
    for (int vv = 0; vv < 8; ++vv) {
      int srow = q0 + 8 * half + vv;
      float val = o[t][vv] / lrow[vv];
      attn_ws[((size_t)b * SEQ + srow) * DIM + hh * HD + d] = (bf16)val;
    }
  }
}

// ---------------------------------------------------------------------------
// Kernel 3: out = attn @ w_out -> f32 d_out section 0.
// grid = (M/64, DIM/64), block = 128.
// ---------------------------------------------------------------------------
__global__ __launch_bounds__(128) void
k_out(const bf16* __restrict__ attn, const bf16* __restrict__ woutT,
      float* __restrict__ out) {
  const int lane = threadIdx.x & 31;
  const int wave = threadIdx.x >> 5;
  const int m0 = blockIdx.x * 64 + wave * 16;
  const int n0 = blockIdx.y * 64;

  v8f acc[4];
#pragma unroll
  for (int t = 0; t < 4; ++t) acc[t] = (v8f)(0.0f);

  for (int k = 0; k < DIM; k += 32) {
    v16bf a = load_frag_a(attn + (size_t)m0 * DIM + k, DIM);
#pragma unroll
    for (int t = 0; t < 4; ++t) {
      v16bf b = load_frag_b(woutT + (size_t)(n0 + 16 * t) * DIM + k, DIM);
      acc[t] = wmma_bf16(a, b, acc[t]);
    }
  }

  const int half = lane >> 4, nl = lane & 15;
#pragma unroll
  for (int t = 0; t < 4; ++t) {
#pragma unroll
    for (int vv = 0; vv < 8; ++vv) {
      int m = m0 + 8 * half + vv;
      out[(size_t)m * DIM + n0 + 16 * t + nl] = acc[t][vv];
    }
  }
}

// ---------------------------------------------------------------------------
extern "C" void kernel_launch(void* const* d_in, const int* in_sizes, int n_in,
                              void* d_out, int out_size, void* d_ws, size_t ws_size,
                              hipStream_t stream) {
  const float* x     = (const float*)d_in[0];
  const float* w_qkv = (const float*)d_in[1];
  const float* w_out = (const float*)d_in[2];

  float* outp  = (float*)d_out;                              // (B,S,D)
  float* out_k = outp  + (size_t)BATCH * SEQ * DIM;          // (B,H,S,HD)
  float* out_v = out_k + (size_t)BATCH * HEADS * SEQ * HD;   // (B,H,S,HD)

  // workspace carve-up (bf16 buffers, 16B aligned)
  char* ws = (char*)d_ws;
  const size_t SZ_X    = (size_t)BATCH * SEQ * DIM * 2;      // 16 MB
  const size_t SZ_WQKV = (size_t)DIM * NQKV * 2;             //  6 MB
  const size_t SZ_WOUT = (size_t)DIM * DIM * 2;              //  2 MB
  const size_t SZ_HSD  = (size_t)BATCH * HEADS * SEQ * HD * 2; // 16 MB

  bf16* xb     = (bf16*)ws;                 ws += SZ_X;
  bf16* wqkvT  = (bf16*)ws;                 ws += SZ_WQKV;
  bf16* woutT  = (bf16*)ws;                 ws += SZ_WOUT;
  bf16* q_ws   = (bf16*)ws;                 ws += SZ_HSD;
  bf16* k_ws   = (bf16*)ws;                 ws += SZ_HSD;
  bf16* v_wsT  = (bf16*)ws;                 ws += SZ_HSD;
  bf16* attn_ws= (bf16*)ws;                 ws += SZ_X;

  const int nx = BATCH * SEQ * DIM;
  k_cvt  <<<(nx + 255) / 256, 256, 0, stream>>>(x, xb, nx);
  k_cvt_t<<<(DIM * NQKV + 255) / 256, 256, 0, stream>>>(w_qkv, wqkvT, DIM, NQKV);
  k_cvt_t<<<(DIM * DIM  + 255) / 256, 256, 0, stream>>>(w_out, woutT, DIM, DIM);

  k_qkv <<<dim3(BATCH * SEQ / 64, NQKV / 64), 128, 0, stream>>>(
      xb, wqkvT, out_k, out_v, q_ws, k_ws, v_wsT);

  k_attn<<<dim3(SEQ / 64, BATCH * HEADS), 128, 0, stream>>>(
      q_ws, k_ws, v_wsT, attn_ws);

  k_out <<<dim3(BATCH * SEQ / 64, DIM / 64), 128, 0, stream>>>(
      attn_ws, woutT, outp);
}
